// DeformableTemporalBlock_13237089206820
// MI455X (gfx1250) — matrix operand order repaired and checked by reading, compiled, not run on hardware
//
#include <hip/hip_runtime.h>

#define M_ 8
#define T_ 4000
#define B_ 128
#define H_ 512
#define KK_ 3
#define EPS_ 1e-8f
#define MTH_ (M_ * T_ * H_)
#define MTB_ (M_ * T_ * B_)

typedef __attribute__((ext_vector_type(16))) __bf16        v16bf;
typedef __attribute__((ext_vector_type(8)))  float         v8f;
typedef __attribute__((ext_vector_type(8)))  unsigned int  v8u;

// ---- f32 -> bf16 (RNE), used only in bandwidth-bound conversion passes ----
__device__ __forceinline__ unsigned pk2bf_u32(float a, float b) {
  unsigned ua = __builtin_bit_cast(unsigned, a);
  unsigned ub = __builtin_bit_cast(unsigned, b);
  ua = (ua + 0x7FFFu + ((ua >> 16) & 1u)) >> 16;
  ub = (ub + 0x7FFFu + ((ub >> 16) & 1u)) & 0xFFFF0000u;
  return (ua & 0xFFFFu) | ub;
}
__device__ __forceinline__ unsigned short f2bf16(float f) {
  unsigned u = __builtin_bit_cast(unsigned, f);
  return (unsigned short)((u + 0x7FFFu + ((u >> 16) & 1u)) >> 16);
}
__device__ __forceinline__ float bflo(unsigned u) { return __builtin_bit_cast(float, u << 16); }
__device__ __forceinline__ float bfhi(unsigned u) { return __builtin_bit_cast(float, u & 0xFFFF0000u); }

// ---- bf16 fragment loads (raw 16B vector loads, ISA lane layouts) ----
// A (16-bit A 16x32): per lane two runs of 8 consecutive K; p = row + ks + 8*g
__device__ __forceinline__ v16bf loadfragA(const unsigned short* __restrict__ p) {
  uint4 q0 = *(const uint4*)(p);         // K+0..7
  uint4 q1 = *(const uint4*)(p + 16);    // K+16..23
  v8u r = {q0.x, q0.y, q0.z, q0.w, q1.x, q1.y, q1.z, q1.w};
  return __builtin_bit_cast(v16bf, r);
}
// B (16-bit B 32x16): per lane two runs of 8 consecutive K; p = col + ks + 8*g
__device__ __forceinline__ v16bf loadfragB(const unsigned short* __restrict__ p) {
  uint4 q0 = *(const uint4*)(p);
  uint4 q1 = *(const uint4*)(p + 16);
  v8u r = {q0.x, q0.y, q0.z, q0.w, q1.x, q1.y, q1.z, q1.w};
  return __builtin_bit_cast(v16bf, r);
}

__device__ __forceinline__ float preluf(float x, float a) { return x >= 0.f ? x : a * x; }

__device__ __forceinline__ float waveReduceSum(float v) {
#pragma unroll
  for (int o = 16; o > 0; o >>= 1) v += __shfl_xor(v, o, 32);
  return v;
}

// ---------------- K0: zero stat accumulators ----------------
__global__ void k0_init(float* st) {
  if (threadIdx.x < 64) st[threadIdx.x] = 0.f;
}

// ---------------- generic f32 -> bf16 conversion (8 elems/thread) ----------------
__global__ __launch_bounds__(256) void k_cvt_bf16(const float* __restrict__ in,
                                                  unsigned* __restrict__ outp, int n8) {
  int i = blockIdx.x * 256 + threadIdx.x;
  if (i >= n8) return;
  const float4* q = (const float4*)(in + i * 8);
  float4 a = q[0], b = q[1];
  uint4 o;
  o.x = pk2bf_u32(a.x, a.y); o.y = pk2bf_u32(a.z, a.w);
  o.z = pk2bf_u32(b.x, b.y); o.w = pk2bf_u32(b.z, b.w);
  ((uint4*)outp)[i] = o;
}

// ---------------- K1: GEMM1 (x @ W1^T) + PReLU + gLN stats ----------------
// Each wave: 64 rows x 32 cols (8 accumulators). Per k-step: 12 b128 loads -> 8 WMMAs.
__global__ __launch_bounds__(256) void k1_gemm1(const unsigned short* __restrict__ xbf,
    const unsigned short* __restrict__ w1bf, const float* __restrict__ prelu1_a,
    float* __restrict__ hbuf, float* __restrict__ sum1, float* __restrict__ ssq1) {
  const int wave = (blockIdx.x << 3) + (threadIdx.x >> 5);   // 8000 waves
  const int lane = threadIdx.x & 31;
  const int g = lane >> 4, r16 = lane & 15;
  const int rt4 = wave >> 4;          // 0..499 (64-row supertile)
  const int cp  = wave & 15;          // 0..15  (32-col supertile)
  const int col0 = cp * 32 + r16;
  const int col1 = col0 + 16;
  const int rowBase = rt4 * 64;
  const float a1 = prelu1_a[0];
  v8f acc0[4] = {}, acc1[4] = {};
#pragma unroll
  for (int ks = 0; ks < B_; ks += 32) {
    // issue ALL fragment loads for this k-step first
    v16bf b0 = loadfragB(w1bf + col0 * B_ + ks + (g << 3));
    v16bf b1 = loadfragB(w1bf + col1 * B_ + ks + (g << 3));
    const unsigned short* ap = xbf + (rowBase + r16) * B_ + ks + (g << 3);
    v16bf af[4];
#pragma unroll
    for (int i = 0; i < 4; ++i) af[i] = loadfragA(ap + i * 16 * B_);
    // then all WMMAs (A fragments reused across both column tiles)
#pragma unroll
    for (int i = 0; i < 4; ++i)
      acc0[i] = __builtin_amdgcn_wmma_f32_16x16x32_bf16(false, af[i], false, b0,
                                                        (short)0, acc0[i], false, false);
#pragma unroll
    for (int i = 0; i < 4; ++i)
      acc1[i] = __builtin_amdgcn_wmma_f32_16x16x32_bf16(false, af[i], false, b1,
                                                        (short)0, acc1[i], false, false);
  }
#pragma unroll
  for (int i = 0; i < 4; ++i) {
    const int tbase = rowBase + 16 * i;          // 16-row tile: within one sample
    float s = 0.f, ss = 0.f;
#pragma unroll
    for (int r = 0; r < 8; ++r) {
      int orow = tbase + r + (g << 3);
      float v0 = preluf(acc0[i][r], a1);
      float v1 = preluf(acc1[i][r], a1);
      hbuf[orow * H_ + col0] = v0;
      hbuf[orow * H_ + col1] = v1;
      s += v0 + v1; ss += v0 * v0 + v1 * v1;
    }
    s = waveReduceSum(s); ss = waveReduceSum(ss);
    if (lane == 0) {
      int m = tbase / T_;
      atomicAdd(&sum1[m], s);
      atomicAdd(&ssq1[m], ss);
    }
  }
}

// ---------------- finalize gLN stats ----------------
__global__ void k_finalize(const float* sum, const float* ssq, float* mean, float* rstd) {
  int m = threadIdx.x;
  if (m < M_) {
    float inv = 1.0f / (float)(T_ * H_);
    float mu  = sum[m] * inv;
    float var = ssq[m] * inv - mu * mu;
    mean[m] = mu;
    rstd[m] = rsqrtf(var + EPS_);
  }
}

// ---------------- K3: apply gLN1 in place (f32, vectorized x4) ----------------
__global__ __launch_bounds__(256) void k3_norm(float* __restrict__ hbuf,
    const float* __restrict__ g1v, const float* __restrict__ b1v,
    const float* __restrict__ mean, const float* __restrict__ rstd) {
  int i4 = (blockIdx.x * 256 + threadIdx.x) * 4;   // grid sized exactly MTH/1024
  int m = i4 / (T_ * H_);
  int h = i4 % H_;
  float mu = mean[m], rs = rstd[m];
  float4 v = *(const float4*)(hbuf + i4);
  float4 gg = *(const float4*)(g1v + h);
  float4 bb = *(const float4*)(b1v + h);
  v.x = (v.x - mu) * rs * gg.x + bb.x;
  v.y = (v.y - mu) * rs * gg.y + bb.y;
  v.z = (v.z - mu) * rs * gg.z + bb.z;
  v.w = (v.w - mu) * rs * gg.w + bb.w;
  *(float4*)(hbuf + i4) = v;
}

// ---------------- K4: offset branch ----------------
__global__ __launch_bounds__(256) void k4_offsets(const float* __restrict__ hcn,
    const float* __restrict__ dw, const float* __restrict__ odc_a,
    const float* __restrict__ pw, const float* __restrict__ opc_a,
    float* __restrict__ offs) {
  const int mt = blockIdx.x;      // m*T + t
  const int t  = mt % T_;
  const int m  = mt / T_;
  const int tid = threadIdx.x;
  int tm1 = (t - 1 < 0)   ? 1      : t - 1;   // reflect pad=1
  int tp1 = (t + 1 >= T_) ? T_ - 2 : t + 1;
  const float* r0 = hcn + (m * T_ + tm1) * H_;
  const float* r1 = hcn + mt * H_;
  const float* r2 = hcn + (m * T_ + tp1) * H_;
  const float oa = odc_a[0];
  float a0 = 0.f, a1 = 0.f, a2 = 0.f;
#pragma unroll
  for (int c = tid; c < H_; c += 256) {
    float d = r0[c] * dw[c * 3 + 0] + r1[c] * dw[c * 3 + 1] + r2[c] * dw[c * 3 + 2];
    d = preluf(d, oa);
    a0 += d * pw[c];
    a1 += d * pw[H_ + c];
    a2 += d * pw[2 * H_ + c];
  }
  a0 = waveReduceSum(a0); a1 = waveReduceSum(a1); a2 = waveReduceSum(a2);
  __shared__ float red[24];
  int w = tid >> 5;
  if ((tid & 31) == 0) { red[w * 3] = a0; red[w * 3 + 1] = a1; red[w * 3 + 2] = a2; }
  __syncthreads();
  if (tid == 0) {
    float s0 = 0, s1 = 0, s2 = 0;
    for (int i = 0; i < 8; ++i) { s0 += red[i * 3]; s1 += red[i * 3 + 1]; s2 += red[i * 3 + 2]; }
    float pa = opc_a[0];
    offs[mt * 3 + 0] = preluf(s0, pa);
    offs[mt * 3 + 1] = preluf(s1, pa);
    offs[mt * 3 + 2] = preluf(s2, pa);
  }
}

// ---------------- K5: deformable depthwise conv + PReLU (store bf16) + gLN stats ----------------
__global__ __launch_bounds__(256) void k5_deform(const float* __restrict__ hcn,
    const float* __restrict__ offs, const float* __restrict__ dww,
    const float* __restrict__ dwb, const float* __restrict__ prelu2_a,
    unsigned short* __restrict__ ybb, float* __restrict__ sum2, float* __restrict__ ssq2) {
  const int mt = blockIdx.x;
  const int t  = mt % T_;
  const int m  = mt / T_;
  const int tid = threadIdx.x;
  float g0[3], g1[3];
  int j0[3], j1[3];
#pragma unroll
  for (int k = 0; k < 3; ++k) {
    float Tpos = (float)t + (float)(2 * k) + offs[mt * 3 + k];
    Tpos = fminf(fmaxf(Tpos, (float)t), (float)(t + 4));   // clip to receptive field
    int U = (int)floorf(Tpos);
    U = min(max(U, 0), T_ + 2);                            // Lp-2 with pad_tot=4
    float Uf = (float)U;
    g0[k] = fmaxf(1.0f - fabsf(Uf - Tpos), 0.0f);
    g1[k] = fmaxf(1.0f - fabsf(Uf + 1.0f - Tpos), 0.0f);
    int a = U - 2, b = U - 1;                              // un-pad (pl = 2)
    if (a < 0) a = -a; else if (a >= T_) a = 2 * T_ - 2 - a;   // reflect
    if (b < 0) b = -b; else if (b >= T_) b = 2 * T_ - 2 - b;
    j0[k] = (m * T_ + a) * H_;
    j1[k] = (m * T_ + b) * H_;
  }
  const float a2v = prelu2_a[0];
  float s = 0.f, ss = 0.f;
#pragma unroll
  for (int h = tid; h < H_; h += 256) {
    float acc = dwb[h];
#pragma unroll
    for (int k = 0; k < 3; ++k)
      acc += (g0[k] * hcn[j0[k] + h] + g1[k] * hcn[j1[k] + h]) * dww[h * 3 + k];
    float v = preluf(acc, a2v);
    ybb[mt * H_ + h] = f2bf16(v);
    s += v; ss += v * v;
  }
  s = waveReduceSum(s); ss = waveReduceSum(ss);
  __shared__ float red[16];
  int w = tid >> 5;
  if ((tid & 31) == 0) { red[w] = s; red[8 + w] = ss; }
  __syncthreads();
  if (tid == 0) {
    float ts = 0, tss = 0;
    for (int i = 0; i < 8; ++i) { ts += red[i]; tss += red[8 + i]; }
    atomicAdd(&sum2[m], ts);
    atomicAdd(&ssq2[m], tss);
  }
}

// ---------------- K6b: gLN2 + re-pack bf16 A-matrix for GEMM2 (8 elems/thread) ----------------
__global__ __launch_bounds__(256) void k6_normcvt(const unsigned short* __restrict__ ybb,
    const float* __restrict__ g2v, const float* __restrict__ b2v,
    const float* __restrict__ mean, const float* __restrict__ rstd,
    unsigned* __restrict__ ynb) {
  int i = blockIdx.x * 256 + threadIdx.x;     // grid sized exactly MTH/2048
  int i8 = i * 8;
  int m = i8 / (T_ * H_);
  int h = i8 % H_;
  float mu = mean[m], rs = rstd[m];
  uint4 q = ((const uint4*)ybb)[i];
  float4 ga = *(const float4*)(g2v + h), gb = *(const float4*)(g2v + h + 4);
  float4 ba = *(const float4*)(b2v + h), bb = *(const float4*)(b2v + h + 4);
  float f0 = (bflo(q.x) - mu) * rs * ga.x + ba.x;
  float f1 = (bfhi(q.x) - mu) * rs * ga.y + ba.y;
  float f2 = (bflo(q.y) - mu) * rs * ga.z + ba.z;
  float f3 = (bfhi(q.y) - mu) * rs * ga.w + ba.w;
  float f4 = (bflo(q.z) - mu) * rs * gb.x + bb.x;
  float f5 = (bfhi(q.z) - mu) * rs * gb.y + bb.y;
  float f6 = (bflo(q.w) - mu) * rs * gb.z + bb.z;
  float f7 = (bfhi(q.w) - mu) * rs * gb.w + bb.w;
  uint4 o;
  o.x = pk2bf_u32(f0, f1); o.y = pk2bf_u32(f2, f3);
  o.z = pk2bf_u32(f4, f5); o.w = pk2bf_u32(f6, f7);
  ((uint4*)ynb)[i] = o;
}

// ---------------- K7: GEMM2 (ynorm @ pw^T) + residual ----------------
// Each wave: 64 rows x 32 cols (8 accumulators); K = 512 (16 steps).
__global__ __launch_bounds__(256) void k7_gemm2(const unsigned short* __restrict__ ynb,
    const unsigned short* __restrict__ pwbf, const float* __restrict__ x,
    float* __restrict__ out) {
  const int wave = (blockIdx.x << 3) + (threadIdx.x >> 5);   // 2000 waves
  const int lane = threadIdx.x & 31;
  const int g = lane >> 4, r16 = lane & 15;
  const int rt4 = wave >> 2;          // 0..499
  const int cp  = wave & 3;           // 0..3 (32-col supertile; B=128)
  const int col0 = cp * 32 + r16;
  const int col1 = col0 + 16;
  const int rowBase = rt4 * 64;
  v8f acc0[4] = {}, acc1[4] = {};
#pragma unroll 4
  for (int ks = 0; ks < H_; ks += 32) {
    v16bf b0 = loadfragB(pwbf + col0 * H_ + ks + (g << 3));
    v16bf b1 = loadfragB(pwbf + col1 * H_ + ks + (g << 3));
    const unsigned short* ap = ynb + (rowBase + r16) * H_ + ks + (g << 3);
    v16bf af[4];
#pragma unroll
    for (int i = 0; i < 4; ++i) af[i] = loadfragA(ap + i * 16 * H_);
#pragma unroll
    for (int i = 0; i < 4; ++i)
      acc0[i] = __builtin_amdgcn_wmma_f32_16x16x32_bf16(false, af[i], false, b0,
                                                        (short)0, acc0[i], false, false);
#pragma unroll
    for (int i = 0; i < 4; ++i)
      acc1[i] = __builtin_amdgcn_wmma_f32_16x16x32_bf16(false, af[i], false, b1,
                                                        (short)0, acc1[i], false, false);
  }
#pragma unroll
  for (int i = 0; i < 4; ++i) {
#pragma unroll
    for (int r = 0; r < 8; ++r) {
      int orow = rowBase + 16 * i + r + (g << 3);
      out[orow * B_ + col0] = acc0[i][r] + x[orow * B_ + col0];   // residual add
      out[orow * B_ + col1] = acc1[i][r] + x[orow * B_ + col1];
    }
  }
}

extern "C" void kernel_launch(void* const* d_in, const int* in_sizes, int n_in,
                              void* d_out, int out_size, void* d_ws, size_t ws_size,
                              hipStream_t stream) {
  const float* x        = (const float*)d_in[0];
  const float* conv1_w  = (const float*)d_in[1];
  const float* prelu1_a = (const float*)d_in[2];
  const float* norm1_g  = (const float*)d_in[3];
  const float* norm1_b  = (const float*)d_in[4];
  const float* off_dw_w = (const float*)d_in[5];
  const float* odc_a    = (const float*)d_in[6];
  const float* off_pw_w = (const float*)d_in[7];
  const float* opc_a    = (const float*)d_in[8];
  const float* dw_w     = (const float*)d_in[9];
  const float* dw_b     = (const float*)d_in[10];
  const float* prelu2_a = (const float*)d_in[11];
  const float* norm2_g  = (const float*)d_in[12];
  const float* norm2_b  = (const float*)d_in[13];
  const float* pw_w     = (const float*)d_in[14];
  float* out = (float*)d_out;

  // workspace layout (all 16B-aligned by construction)
  float*          hbuf = (float*)d_ws;                      // [M,T,H] f32
  unsigned short* ybb  = (unsigned short*)(hbuf + MTH_);    // [M,T,H] bf16 (post-prelu y)
  unsigned short* ynb  = ybb + MTH_;                        // [M,T,H] bf16 (gLN2 applied)
  unsigned short* xbf  = ynb + MTH_;                        // [M,T,B] bf16
  unsigned short* w1bf = xbf + MTB_;                        // [H,B]   bf16
  unsigned short* pwbf = w1bf + H_ * B_;                    // [B,H]   bf16
  float* offs = (float*)(pwbf + B_ * H_);                   // [M,T,3] f32
  float* st   = offs + M_ * T_ * KK_;                       // 64 floats of stats
  float* sum1 = st +  0; float* ssq1 = st +  8; float* mean1 = st + 16; float* rstd1 = st + 24;
  float* sum2 = st + 32; float* ssq2 = st + 40; float* mean2 = st + 48; float* rstd2 = st + 56;

  hipLaunchKernelGGL(k0_init, dim3(1), dim3(64), 0, stream, st);
  // pre-convert activations/weights to bf16 (memory-bound, one-shot)
  hipLaunchKernelGGL(k_cvt_bf16, dim3(MTB_ / 8 / 256), dim3(256), 0, stream,
                     x, (unsigned*)xbf, MTB_ / 8);
  hipLaunchKernelGGL(k_cvt_bf16, dim3((H_ * B_ / 8 + 255) / 256), dim3(256), 0, stream,
                     conv1_w, (unsigned*)w1bf, H_ * B_ / 8);
  hipLaunchKernelGGL(k_cvt_bf16, dim3((B_ * H_ / 8 + 255) / 256), dim3(256), 0, stream,
                     pw_w, (unsigned*)pwbf, B_ * H_ / 8);

  hipLaunchKernelGGL(k1_gemm1, dim3((M_ * T_ / 64) * (H_ / 32) / 8), dim3(256), 0, stream,
                     xbf, w1bf, prelu1_a, hbuf, sum1, ssq1);
  hipLaunchKernelGGL(k_finalize, dim3(1), dim3(32), 0, stream, sum1, ssq1, mean1, rstd1);
  hipLaunchKernelGGL(k3_norm, dim3(MTH_ / 1024), dim3(256), 0, stream,
                     hbuf, norm1_g, norm1_b, mean1, rstd1);
  hipLaunchKernelGGL(k4_offsets, dim3(M_ * T_), dim3(256), 0, stream,
                     hbuf, off_dw_w, odc_a, off_pw_w, opc_a, offs);
  hipLaunchKernelGGL(k5_deform, dim3(M_ * T_), dim3(256), 0, stream,
                     hbuf, offs, dw_w, dw_b, prelu2_a, ybb, sum2, ssq2);
  hipLaunchKernelGGL(k_finalize, dim3(1), dim3(32), 0, stream, sum2, ssq2, mean2, rstd2);
  hipLaunchKernelGGL(k6_normcvt, dim3(MTH_ / 2048), dim3(256), 0, stream,
                     ybb, norm2_g, norm2_b, mean2, rstd2, (unsigned*)ynb);
  hipLaunchKernelGGL(k7_gemm2, dim3((M_ * T_ / 64) * (B_ / 32) / 8), dim3(256), 0, stream,
                     ynb, pwbf, x, out);
}
